// ScaledDotProductAttention_50740743635464
// MI455X (gfx1250) — matrix operand order repaired and checked by reading, not compiled, not run on hardware
//
#include <hip/hip_runtime.h>
#include <hip/hip_bf16.h>

// ---------------------------------------------------------------------------
// Flash attention for B=4,H=16,S=2048,D=128 (fp32 in, fp32 out) on gfx1250.
// bf16 WMMA (v_wmma_f32_16x16x32_bf16) for QK^T and A*V, fp32 softmax.
// Double-buffered LDS tiles, register-pipelined global loads, preloaded
// WMMA B-fragments, DPP row_ror rotation all-reduce for softmax.
// Outputs: QKV [B,H,S,D] followed by score [B,H,S] (column sums of A).
// ---------------------------------------------------------------------------

typedef __attribute__((ext_vector_type(16))) __bf16 v16bf;
typedef __attribute__((ext_vector_type(8)))  __bf16 v8bf;
typedef __attribute__((ext_vector_type(8)))  float  v8f;
typedef __attribute__((ext_vector_type(4)))  float  v4f;

#define ATT_B   4
#define ATT_H   16
#define ATT_S   2048
#define ATT_D   128
#define BM      128          // queries per workgroup (8 waves x 16)
#define BN      32           // keys per iteration
#define NKB     (ATT_S / BN) // 64 key blocks
#define INV_SCALE 0.08838834764831845f   // 1/sqrt(128)

__device__ __forceinline__ v8f wmma_bf16(v16bf a, v16bf b, v8f c) {
    // D = A(16x32 bf16) x B(32x16 bf16) + C(16x16 f32)
    return __builtin_amdgcn_wmma_f32_16x16x32_bf16(
        /*neg_a=*/false, a, /*neg_b=*/false, b,
        /*c_mod=*/(short)0, c, /*reuse_a=*/false, /*reuse_b=*/false);
}

// DPP row_ror:n within a 16-lane row (dpp_ctrl = 0x120 + n)
template <int N>
__device__ __forceinline__ float dpp_ror(float v) {
    return __int_as_float(__builtin_amdgcn_update_dpp(
        0, __float_as_int(v), 0x120 + N, 0xf, 0xf, true));
}
// rotation-based all-reduce over the 16-lane row (valid for commutative ops)
__device__ __forceinline__ float rowmax16(float t) {
    t = fmaxf(t, dpp_ror<1>(t));
    t = fmaxf(t, dpp_ror<2>(t));
    t = fmaxf(t, dpp_ror<4>(t));
    t = fmaxf(t, dpp_ror<8>(t));
    return t;
}
__device__ __forceinline__ float rowsum16(float t) {
    t += dpp_ror<1>(t);
    t += dpp_ror<2>(t);
    t += dpp_ror<4>(t);
    t += dpp_ror<8>(t);
    return t;
}

__global__ void __launch_bounds__(256)
attn_fwd_kernel(const float* __restrict__ Q,
                const float* __restrict__ K,
                const float* __restrict__ V,
                float* __restrict__ out)   // [B*H*S*D] QKV then [B*H*S] score
{
    __shared__ __align__(128) __bf16 Kt[2][BN][ATT_D];   // 16 KB, (key, d)
    __shared__ __align__(128) __bf16 Vt[2][ATT_D][BN];   // 16 KB, transposed (d, key)
    __shared__ __align__(128) __bf16 Pt[8][16][BN];      // 8 KB, per-wave P tile
    __shared__ __align__(16)  float  sAcc[BN];           // per-block score partials

    const int tid  = threadIdx.x;
    const int wave = tid >> 5;
    const int lane = tid & 31;
    const int hl   = lane >> 4;    // lane half (0: lanes 0-15, 1: lanes 16-31)
    const int ln   = lane & 15;

    const int bh = blockIdx.y;                 // b*H + h
    const int q0 = blockIdx.x * BM + wave * 16;

    const size_t tens = (size_t)bh * ATT_S * ATT_D;
    const float* Qg = Q + tens;
    const float* Kg = K + tens;
    const float* Vg = V + tens;
    float*       Og = out + tens;
    float*       Sg = out + (size_t)ATT_B * ATT_H * ATT_S * ATT_D + (size_t)bh * ATT_S;

    // ---- Q tile -> bf16 A-matrix fragments, pre-scaled by 1/sqrt(D) ----------
    // lane row M = ln; elements 0..7  <-> d = 32c + 16*hl + e
    //               elements 8..15 <-> d = 32c + 16*hl + 8 + e
    v16bf qa[4];
    {
        const float* Qrow = Qg + (size_t)(q0 + ln) * ATT_D;
#pragma unroll
        for (int c = 0; c < 4; ++c) {
#pragma unroll
            for (int e = 0; e < 8; ++e) {
                qa[c][e]     = (__bf16)(Qrow[32 * c + 16 * hl + e] * INV_SCALE);
                qa[c][8 + e] = (__bf16)(Qrow[32 * c + 16 * hl + 8 + e] * INV_SCALE);
            }
        }
    }

    // ---- running state: rows r (lanes<16 -> M=r, lanes>=16 -> M=r+8) ---------
    v8f Oacc[8];                   // 16x128 output tile: 8 col-groups x 8 rows
#pragma unroll
    for (int dg = 0; dg < 8; ++dg) Oacc[dg] = v8f{};
    float mrun[8], lrun[8];
#pragma unroll
    for (int r = 0; r < 8; ++r) { mrun[r] = -INFINITY; lrun[r] = 0.0f; }

    // cooperative-load coordinates: 16 floats per thread (contiguous in d)
    const int crow = tid >> 3;           // 0..31
    const int ccol = (tid & 7) * 16;     // 0..112
    const float* kBase = Kg + (size_t)crow * ATT_D + ccol;
    const float* vBase = Vg + (size_t)crow * ATT_D + ccol;

    // ======================= pass 1: online softmax + O =======================
    v4f kreg[4], vreg[4];
#pragma unroll
    for (int j = 0; j < 4; ++j) {        // preload tile 0 into registers
        kreg[j] = *(const v4f*)(kBase + 4 * j);
        vreg[j] = *(const v4f*)(vBase + 4 * j);
    }

    for (int kb = 0; kb < NKB; ++kb) {
        const int buf = kb & 1;

        // ---- convert current registers -> LDS tile (write buffer) ------------
        {
            v16bf kv;
#pragma unroll
            for (int j = 0; j < 4; ++j)
#pragma unroll
                for (int e = 0; e < 4; ++e) {
                    kv[4 * j + e] = (__bf16)kreg[j][e];
                    Vt[buf][ccol + 4 * j + e][crow] = (__bf16)vreg[j][e];
                }
            *(v16bf*)&Kt[buf][crow][ccol] = kv;    // one 32B LDS store
        }
        // ---- issue next tile's global loads (latency hidden by compute) ------
        if (kb + 1 < NKB) {
            const float* kSrc = kBase + (size_t)(kb + 1) * BN * ATT_D;
            const float* vSrc = vBase + (size_t)(kb + 1) * BN * ATT_D;
#pragma unroll
            for (int j = 0; j < 4; ++j) {
                kreg[j] = *(const v4f*)(kSrc + 4 * j);
                vreg[j] = *(const v4f*)(vSrc + 4 * j);
            }
        }
        __syncthreads();

        // ---- S = Q * K^T: preload all 8 B-fragments, then 8 WMMAs ------------
        v16bf bfrag[8];
#pragma unroll
        for (int c = 0; c < 4; ++c) {
            // B 32x16 layout: lane col N = ln, rows split by half
            bfrag[2 * c]     = *(const v16bf*)&Kt[buf][ln][c * 32 + hl * 16];
            bfrag[2 * c + 1] = *(const v16bf*)&Kt[buf][16 + ln][c * 32 + hl * 16];
        }
        v8f s0 = v8f{}, s1 = v8f{};
#pragma unroll
        for (int c = 0; c < 4; ++c) {
            s0 = wmma_bf16(qa[c], bfrag[2 * c], s0);
            s1 = wmma_bf16(qa[c], bfrag[2 * c + 1], s1);
        }

        // ---- online softmax over rows (DPP rotation reductions) ---------------
        float corr[8];
#pragma unroll
        for (int r = 0; r < 8; ++r) {
            float a0 = s0[r];
            float a1 = s1[r];
            float t  = rowmax16(fmaxf(a0, a1));
            float mn = fmaxf(mrun[r], t);
            corr[r]  = __expf(mrun[r] - mn);
            mrun[r]  = mn;
            float p0 = __expf(a0 - mn);
            float p1 = __expf(a1 - mn);
            float rs = rowsum16(p0 + p1);
            lrun[r]  = lrun[r] * corr[r] + rs;
            // stash P (un-normalized) for the A*V matmul
            int row = r + 8 * hl;
            Pt[wave][row][ln]      = (__bf16)p0;
            Pt[wave][row][16 + ln] = (__bf16)p1;
        }
#pragma unroll
        for (int dg = 0; dg < 8; ++dg)
#pragma unroll
            for (int r = 0; r < 8; ++r) Oacc[dg][r] *= corr[r];

        __builtin_amdgcn_wave_barrier();   // wave-internal LDS reshape fence

        // ---- reload P in A-matrix layout, preload V fragments ----------------
        v8bf lo = *(const v8bf*)&Pt[wave][ln][16 * hl];
        v8bf hi = *(const v8bf*)&Pt[wave][ln][16 * hl + 8];
        v16bf pa;
#pragma unroll
        for (int e = 0; e < 8; ++e) { pa[e] = lo[e]; pa[8 + e] = hi[e]; }

        v16bf vfrag[8];
#pragma unroll
        for (int dg = 0; dg < 8; ++dg)
            vfrag[dg] = *(const v16bf*)&Vt[buf][dg * 16 + ln][16 * hl];

        // ---- O += P * V -------------------------------------------------------
#pragma unroll
        for (int dg = 0; dg < 8; ++dg)
            Oacc[dg] = wmma_bf16(pa, vfrag[dg], Oacc[dg]);
    }

    // ---- finalize O ----------------------------------------------------------
    float invl[8];
#pragma unroll
    for (int r = 0; r < 8; ++r) invl[r] = 1.0f / lrun[r];
#pragma unroll
    for (int dg = 0; dg < 8; ++dg)
#pragma unroll
        for (int r = 0; r < 8; ++r) {
            int q = q0 + r + 8 * hl;
            Og[(size_t)q * ATT_D + dg * 16 + ln] = Oacc[dg][r] * invl[r];
        }

    // ======================= pass 2: score (column sums) ======================
#pragma unroll
    for (int j = 0; j < 4; ++j) kreg[j] = *(const v4f*)(kBase + 4 * j);

    for (int kb = 0; kb < NKB; ++kb) {
        const int buf = kb & 1;
        {
            v16bf kv;
#pragma unroll
            for (int j = 0; j < 4; ++j)
#pragma unroll
                for (int e = 0; e < 4; ++e) kv[4 * j + e] = (__bf16)kreg[j][e];
            *(v16bf*)&Kt[buf][crow][ccol] = kv;
        }
        if (kb + 1 < NKB) {
            const float* kSrc = kBase + (size_t)(kb + 1) * BN * ATT_D;
#pragma unroll
            for (int j = 0; j < 4; ++j) kreg[j] = *(const v4f*)(kSrc + 4 * j);
        }
        if (tid < BN) sAcc[tid] = 0.0f;
        __syncthreads();

        v16bf bfrag[8];
#pragma unroll
        for (int c = 0; c < 4; ++c) {
            bfrag[2 * c]     = *(const v16bf*)&Kt[buf][ln][c * 32 + hl * 16];
            bfrag[2 * c + 1] = *(const v16bf*)&Kt[buf][16 + ln][c * 32 + hl * 16];
        }
        v8f s0 = v8f{}, s1 = v8f{};
#pragma unroll
        for (int c = 0; c < 4; ++c) {
            s0 = wmma_bf16(qa[c], bfrag[2 * c], s0);
            s1 = wmma_bf16(qa[c], bfrag[2 * c + 1], s1);
        }
        float cs0 = 0.0f, cs1 = 0.0f;
#pragma unroll
        for (int r = 0; r < 8; ++r) {
            cs0 += __expf(s0[r] - mrun[r]) * invl[r];
            cs1 += __expf(s1[r] - mrun[r]) * invl[r];
        }
        cs0 += __shfl_xor(cs0, 16, 32);   // add rows from the other lane half
        cs1 += __shfl_xor(cs1, 16, 32);
        if (hl == 0) {
            atomicAdd(&sAcc[ln], cs0);
            atomicAdd(&sAcc[16 + ln], cs1);
        }
        __syncthreads();
        if (tid < BN) atomicAdd(&Sg[kb * BN + tid], sAcc[tid]);
    }
}

__global__ void zero_score_kernel(float* p, int n) {
    int i = blockIdx.x * blockDim.x + threadIdx.x;
    if (i < n) p[i] = 0.0f;
}

extern "C" void kernel_launch(void* const* d_in, const int* in_sizes, int n_in,
                              void* d_out, int out_size, void* d_ws, size_t ws_size,
                              hipStream_t stream) {
    const float* Q = (const float*)d_in[0];
    const float* K = (const float*)d_in[1];
    const float* V = (const float*)d_in[2];
    float* out = (float*)d_out;

    // zero the score region (atomically accumulated)
    const int nScore = ATT_B * ATT_H * ATT_S;
    float* score = out + (size_t)ATT_B * ATT_H * ATT_S * ATT_D;
    zero_score_kernel<<<(nScore + 255) / 256, 256, 0, stream>>>(score, nScore);

    dim3 grid(ATT_S / BM, ATT_B * ATT_H);
    attn_fwd_kernel<<<grid, 256, 0, stream>>>(Q, K, V, out);
}
